// MultiHeadedAttentionWithGate_30520037605936
// MI455X (gfx1250) — compile-verified, matrix operand order checked by reading, not run on hardware
//
#include <hip/hip_runtime.h>
#include <hip/hip_bf16.h>

// MI455X (gfx1250) fused multi-head neighbor attention with gate.
//   B=64, A=128, NEI=10, D_MODEL=256, HEADS=8, D_K=32, K-dim=512
// Strategy: one pass over the 167.8MB input; bf16 WMMA (16x16x32, f32 acc)
// for the 3 big projections + Q; flat-index-space attention/pool/gate
// (the reference's reshapes are raw, so each group g=h*128+a' is a
// contiguous 320-float block of the projected rows). Ungated attention +
// gate logits spill to workspace (~8.6MB, L2-resident); a small second
// kernel applies the softmax-over-heads gate.

#define HEADS   8
#define DMODEL  256
#define DK      32
#define NNEI    10
#define BATCH   64
#define NATOM   128
#define KDIM    512        // 2*D_MODEL
#define ROWS    80         // multihead rows per tile (= 64 groups exactly)
#define GRPS    64
#define NTILES  16         // 1280 / 80
#define XSTRIDE 520        // bf16 LDS stride for X tile (bank rotate)
#define QSTRIDE 264
#define PSTRIDE 260        // f32 LDS stride for projection tile

typedef __attribute__((ext_vector_type(16))) __bf16 v16bf;
typedef __attribute__((ext_vector_type(8)))  float  v8f;

// Load one WMMA 16-bit fragment for this lane from a 32-wide K slice.
// ISA layout (A 16x32 / B 32x16 mirror): lane holds two contiguous 8-half
// chunks at K offsets c0 and 16+c0, c0 = (lane<16 ? 0 : 8).
__device__ __forceinline__ v16bf load_frag(const __bf16* p, int c0) {
  union { v16bf v; uint4 q[2]; } f;
  f.q[0] = *reinterpret_cast<const uint4*>(p + c0);
  f.q[1] = *reinterpret_cast<const uint4*>(p + 16 + c0);
  return f.v;
}

// [80x512 bf16 LDS] x [512x256 via Wt(=W^T) 256x512 bf16 global] -> 80x256 f32 LDS
__device__ __forceinline__ void gemm_mh(const __bf16* Xs, const __bf16* __restrict__ Wt,
                                        const float* __restrict__ bias, float* Out,
                                        int wave, int lane) {
  const int c0  = (lane < 16) ? 0 : 8;
  const int m16 = lane & 15;
  const int rhi = (lane >> 4) << 3;
  for (int tile = wave; tile < 80; tile += 8) {      // 5 m-tiles x 16 n-tiles
    const int mt = tile / 16, nt = tile & 15;
    v8f acc = {};
    const __bf16* arow = Xs + (mt * 16 + m16) * XSTRIDE;
    const __bf16* brow = Wt + (size_t)(nt * 16 + m16) * KDIM;
#pragma unroll 4
    for (int kk = 0; kk < 16; ++kk) {                // K = 512 = 16 x 32
      v16bf a = load_frag(arow + kk * 32, c0);
      v16bf b = load_frag(brow + kk * 32, c0);
      acc = __builtin_amdgcn_wmma_f32_16x16x32_bf16(false, a, false, b,
                                                    (short)0, acc, false, false);
    }
    const int col = nt * 16 + m16;
    const float bv = bias[col];
#pragma unroll
    for (int i = 0; i < 8; ++i)                      // D: VGPR i -> row i + rhi
      Out[(mt * 16 + rhi + i) * PSTRIDE + col] = acc[i] + bv;
  }
}

// [8x256 bf16 LDS, zero padded to 16 rows] x [256x256] -> 8x256 f32 LDS
__device__ __forceinline__ void gemm_q(const __bf16* Xqs, const __bf16* __restrict__ Wt,
                                       const float* __restrict__ bias, float* Qf,
                                       int wave, int lane) {
  const int c0  = (lane < 16) ? 0 : 8;
  const int m16 = lane & 15;
  const int rhi = (lane >> 4) << 3;
  for (int nt = wave; nt < 16; nt += 8) {
    v8f acc = {};
    const __bf16* arow = Xqs + m16 * QSTRIDE;
    const __bf16* brow = Wt + (size_t)(nt * 16 + m16) * DMODEL;
#pragma unroll
    for (int kk = 0; kk < 8; ++kk) {                 // K = 256 = 8 x 32
      v16bf a = load_frag(arow + kk * 32, c0);
      v16bf b = load_frag(brow + kk * 32, c0);
      acc = __builtin_amdgcn_wmma_f32_16x16x32_bf16(false, a, false, b,
                                                    (short)0, acc, false, false);
    }
    const int col = nt * 16 + m16;
    const float bv = bias[col];
#pragma unroll
    for (int i = 0; i < 8; ++i) {
      const int r = rhi + i;
      if (r < 8) Qf[r * PSTRIDE + col] = acc[i] + bv;
    }
  }
}

__global__ void __launch_bounds__(256) prep_wt(const float* __restrict__ W,
                                               __bf16* __restrict__ Wt, int K, int N) {
  int i = blockIdx.x * 256 + threadIdx.x;            // Wt[n][k] = W[k][n]
  if (i < K * N) {
    int n = i / K, k = i - n * K;
    Wt[i] = (__bf16)W[(size_t)k * N + n];
  }
}

__global__ void __launch_bounds__(256) fused_attn(
    const float* __restrict__ xmh, const float* __restrict__ xq,
    const __bf16* __restrict__ WqT, const __bf16* __restrict__ WkT,
    const __bf16* __restrict__ WvT, const __bf16* __restrict__ WamT,
    const float* __restrict__ bq, const float* __restrict__ bk,
    const float* __restrict__ bv, const float* __restrict__ bam,
    const float* __restrict__ Wg, const float* __restrict__ bg,
    float* __restrict__ wsAtt, float* __restrict__ wsLogit)
{
  extern __shared__ char smem[];
  __bf16* Xs  = (__bf16*)smem;                       // [80][520] bf16 input tile
  __bf16* Xqs = Xs + ROWS * XSTRIDE;                 // [16][264] bf16 query rows
  float*  Qf  = (float*)(Xqs + 16 * QSTRIDE);        // [8][260]  q projection
  float*  Pj  = Qf + 8 * PSTRIDE;                    // [80][260] reused: AM -> K -> V
  float*  sc  = Pj + ROWS * PSTRIDE;                 // [64][12]  scores / probs
  float*  gl  = sc + GRPS * 12;                      // [64]      gate logits

  const int tid  = threadIdx.x;
  const int lane = tid & 31;
  const int wave = tid >> 5;
  const int rt   = blockIdx.x;                       // row tile 0..15
  const int b    = blockIdx.y;                       // batch
  const int G0   = rt * GRPS;                        // first group in this b

  // ---- stage multihead rows [rt*80, rt*80+80) x 512 as bf16 ----
  const float* xsrc = xmh + ((size_t)b * 1280 + (size_t)rt * ROWS) * KDIM;
  for (int i = tid; i < ROWS * (KDIM / 4); i += 256) {
    int r = i >> 7, c = i & 127;
    float4 v = reinterpret_cast<const float4*>(xsrc)[(size_t)r * (KDIM / 4) + c];
    __bf16* dst = Xs + r * XSTRIDE + c * 4;
    dst[0] = (__bf16)v.x; dst[1] = (__bf16)v.y;
    dst[2] = (__bf16)v.z; dst[3] = (__bf16)v.w;
  }
  // ---- stage query rows [rt*8, rt*8+8), zero-pad to 16 ----
  const float* qsrc = xq + ((size_t)b * NATOM + (size_t)rt * 8) * DMODEL;
  for (int i = tid; i < 16 * DMODEL; i += 256) {
    int r = i >> 8, c = i & 255;
    Xqs[r * QSTRIDE + c] = (r < 8) ? (__bf16)qsrc[(size_t)r * DMODEL + c] : (__bf16)0.f;
  }
  if (tid < GRPS) gl[tid] = bg[0];
  __syncthreads();

  // ---- Q projection ----
  gemm_q(Xqs, WqT, bq, Qf, wave, lane);

  // ---- gate: cur_atom (Wg[0:32]) + neighbor-mean (Wg[64:128]) in exact f32 ----
  const float* xqflat = xq + (size_t)b * (NATOM * DMODEL);
  for (int i = tid; i < GRPS * DK; i += 256) {
    int lg = i >> 5, d = i & 31;
    atomicAdd(&gl[lg], xqflat[(size_t)(G0 + lg) * DK + d] * Wg[d]);
  }
  const float* xmflat = xmh + (size_t)b * (1280 * KDIM);
  for (int i = tid; i < GRPS * 64; i += 256) {
    int lg = i >> 6, d = i & 63;
    const float* p = xmflat + (size_t)(G0 + lg) * 640 + d;
    float s = 0.f;
#pragma unroll
    for (int n = 0; n < NNEI; ++n) s += p[n * 64];
    atomicAdd(&gl[lg], (s * 0.1f) * Wg[64 + d]);
  }
  __syncthreads();

  // ---- AM projection -> element-max gate contribution (Wg[32:64]) ----
  gemm_mh(Xs, WamT, bam, Pj, wave, lane);
  __syncthreads();
  for (int i = tid; i < GRPS * DK; i += 256) {
    int lg = i >> 5, d = i & 31;
    float m = -3.4e38f;
#pragma unroll
    for (int n = 0; n < NNEI; ++n) {
      int lf = lg * 320 + n * 32 + d;
      m = fmaxf(m, Pj[(lf >> 8) * PSTRIDE + (lf & 255)]);
    }
    atomicAdd(&gl[lg], m * Wg[32 + d]);
  }
  __syncthreads();

  // ---- K projection -> scores ----
  gemm_mh(Xs, WkT, bk, Pj, wave, lane);
  __syncthreads();
  for (int i = tid; i < GRPS * NNEI; i += 256) {
    int lg = i / NNEI, n = i - lg * NNEI;
    const float* q = Qf + (lg >> 3) * PSTRIDE + (lg & 7) * DK;
    float s = 0.f;
#pragma unroll
    for (int d = 0; d < DK; ++d) {
      int lf = lg * 320 + n * 32 + d;
      s += q[d] * Pj[(lf >> 8) * PSTRIDE + (lf & 255)];
    }
    sc[lg * 12 + n] = s;
  }
  __syncthreads();
  if (tid < GRPS) {                                  // softmax over 10 neighbors
    float m = -3.4e38f;
    for (int n = 0; n < NNEI; ++n) m = fmaxf(m, sc[tid * 12 + n]);
    float sum = 0.f;
    for (int n = 0; n < NNEI; ++n) {
      float e = expf(sc[tid * 12 + n] - m);
      sc[tid * 12 + n] = e; sum += e;
    }
    float inv = 1.f / sum;
    for (int n = 0; n < NNEI; ++n) sc[tid * 12 + n] *= inv;
  }
  __syncthreads();

  // ---- V projection -> weighted sum, store ungated attention ----
  gemm_mh(Xs, WvT, bv, Pj, wave, lane);
  __syncthreads();
  for (int i = tid; i < GRPS * DK; i += 256) {
    int lg = i >> 5, d = i & 31;
    float acc = 0.f;
#pragma unroll
    for (int n = 0; n < NNEI; ++n) {
      int lf = lg * 320 + n * 32 + d;
      acc += sc[lg * 12 + n] * Pj[(lf >> 8) * PSTRIDE + (lf & 255)];
    }
    wsAtt[(size_t)b * 32768 + (size_t)(G0 + lg) * DK + d] = acc;
  }
  if (tid < GRPS) wsLogit[(size_t)b * 1024 + G0 + tid] = gl[tid];
}

// softmax-over-heads gate + final scaled write (output flat order == group order)
__global__ void __launch_bounds__(256) gate_finalize(
    const float* __restrict__ wsAtt, const float* __restrict__ wsLogit,
    float* __restrict__ out)
{
  const int b  = blockIdx.x >> 7;
  const int ap = blockIdx.x & 127;
  const int h  = threadIdx.x >> 5;
  const int d  = threadIdx.x & 31;
  const float* lg = wsLogit + (size_t)b * 1024 + ap;
  float m = -3.4e38f;
#pragma unroll
  for (int hh = 0; hh < HEADS; ++hh) m = fmaxf(m, lg[hh * NATOM]);
  float sum = 0.f;
#pragma unroll
  for (int hh = 0; hh < HEADS; ++hh) sum += expf(lg[hh * NATOM] - m);
  const float p = expf(lg[h * NATOM] - m) / sum;
  const size_t idx = (size_t)b * 32768 + (size_t)(h * NATOM + ap) * DK + d;
  out[idx] = p * wsAtt[idx];
}

extern "C" void kernel_launch(void* const* d_in, const int* in_sizes, int n_in,
                              void* d_out, int out_size, void* d_ws, size_t ws_size,
                              hipStream_t stream) {
  (void)in_sizes; (void)n_in; (void)out_size; (void)ws_size;
  const float* xmh = (const float*)d_in[0];
  const float* xq  = (const float*)d_in[1];
  const float* Wq  = (const float*)d_in[2];
  const float* bq  = (const float*)d_in[3];
  const float* Wk  = (const float*)d_in[4];
  const float* bk  = (const float*)d_in[5];
  const float* Wv  = (const float*)d_in[6];
  const float* bv  = (const float*)d_in[7];
  const float* Wam = (const float*)d_in[8];
  const float* bam = (const float*)d_in[9];
  const float* Wg  = (const float*)d_in[10];
  const float* bg  = (const float*)d_in[11];

  char* ws = (char*)d_ws;                 // bf16 W^T blocks, then f32 scratch
  __bf16* WqT  = (__bf16*)(ws + 0);       // 256x256 -> 131072 B
  __bf16* WkT  = (__bf16*)(ws + 131072);  // 256x512 -> 262144 B
  __bf16* WvT  = (__bf16*)(ws + 393216);
  __bf16* WamT = (__bf16*)(ws + 655360);
  float*  wsAtt   = (float*)(ws + 917504);            // 64*32768 f32 = 8 MB
  float*  wsLogit = (float*)(ws + 917504 + 8388608);  // 64*1024  f32 = 256 KB

  prep_wt<<<(65536  + 255) / 256, 256, 0, stream>>>(Wq,  WqT,  256, 256);
  prep_wt<<<(131072 + 255) / 256, 256, 0, stream>>>(Wk,  WkT,  512, 256);
  prep_wt<<<(131072 + 255) / 256, 256, 0, stream>>>(Wv,  WvT,  512, 256);
  prep_wt<<<(131072 + 255) / 256, 256, 0, stream>>>(Wam, WamT, 512, 256);

  const size_t smem = (size_t)(ROWS * XSTRIDE + 16 * QSTRIDE) * sizeof(__bf16)
                    + (size_t)(8 * PSTRIDE + ROWS * PSTRIDE + GRPS * 12 + GRPS) * sizeof(float);
  dim3 grid(NTILES, BATCH);               // 16 x 64 = 1024 workgroups, 8 waves each
  fused_attn<<<grid, 256, smem, stream>>>(xmh, xq, WqT, WkT, WvT, WamT,
                                          bq, bk, bv, bam, Wg, bg, wsAtt, wsLogit);
  gate_finalize<<<BATCH * NATOM, 256, 0, stream>>>(wsAtt, wsLogit, (float*)d_out);
}
